// GraphConvolutionBlock_43920335569181
// MI455X (gfx1250) — matrix operand order repaired and checked by reading, not compiled, not run on hardware
//
#include <hip/hip_runtime.h>
#include <math.h>

// ---------------------------------------------------------------------------
// CGConv GNN block on MI455X (gfx1250, wave32).
// All GEMMs via v_wmma_f32_16x16x32_bf16 (bf16 A/B, f32 accumulate).
//
// Data-movement strategy (23.3 TB/s HBM, 192 MB L2):
//  * edge_attr (819 MB) is streamed ONCE with non-temporal loads so it never
//    evicts the reusable state from L2.
//  * ea is materialized as packed bf16 [E,32] (102 MB, L2-resident): each WMMA
//    A-fragment is then two b128 loads with zero cvt ops in the edge loop.
//  * h has an f32 master (atomic scatter target + residual) and a 3.2 MB bf16
//    mirror used for the per-edge gathers; a tiny convert kernel refreshes the
//    mirror between layers.
//  * CGConv keeps all 12 weight B-fragments resident in VGPRs across a
//    grid-stride loop over 100K edge tiles; dst indices for the epilogue are
//    shared via ds_bpermute (__shfl) instead of re-loading from memory.
// ---------------------------------------------------------------------------

typedef __attribute__((ext_vector_type(16))) __bf16 v16bf;
typedef __attribute__((ext_vector_type(8)))  __bf16 v8bf;
typedef __attribute__((ext_vector_type(8)))  float  v8f;
typedef __attribute__((ext_vector_type(4)))  float  v4f;

#define EPSV  1e-5f
#define SLOPE 0.1f
#define HID   32

static __device__ __forceinline__ v4f ld4(const float* p) {
  return *(const v4f*)p;
}
static __device__ __forceinline__ v4f ld4nt(const float* p) {
  return __builtin_nontemporal_load((const v4f*)p);  // TH=NT: stream-once data
}

static __device__ __forceinline__ void pack8(v16bf& a, int base, v4f x, v4f y) {
#pragma unroll
  for (int i = 0; i < 4; ++i) {
    a[base + i]     = (__bf16)x[i];
    a[base + 4 + i] = (__bf16)y[i];
  }
}

// A fragment 16x32 bf16 (MxK) from an f32 row, ISA 05_wmma layout:
// lane = (hf = lane>>4, m = lane&15); lane supplies row m with
// K in {kbase + hf*8 .. +7} (VGPR0-3) and {kbase + 16 + hf*8 .. +7} (VGPR4-7).
template <bool NT>
static __device__ __forceinline__ v16bf load_a_row(const float* __restrict__ row,
                                                   int kbase, int hf) {
  const float* p0 = row + kbase + hf * 8;
  const float* p1 = p0 + 16;
  v4f a0, a1, a2, a3;
  if (NT) { a0 = ld4nt(p0); a1 = ld4nt(p0 + 4); a2 = ld4nt(p1); a3 = ld4nt(p1 + 4); }
  else    { a0 = ld4(p0);   a1 = ld4(p0 + 4);   a2 = ld4(p1);   a3 = ld4(p1 + 4);   }
  v16bf a;
  pack8(a, 0, a0, a1);
  pack8(a, 8, a2, a3);
  return a;
}

// Same A fragment from a pre-packed bf16 row (natural k order, 64 B/row):
// two 16-byte loads, no conversions.
static __device__ __forceinline__ v16bf load_a_row_bf16(const __bf16* __restrict__ row,
                                                        int hf) {
  v8bf lo = *(const v8bf*)(row + hf * 8);        // k = hf*8 .. +7
  v8bf hi = *(const v8bf*)(row + 16 + hf * 8);   // k = 16 + hf*8 .. +7
  v16bf a;
#pragma unroll
  for (int i = 0; i < 8; ++i) { a[i] = lo[i]; a[8 + i] = hi[i]; }
  return a;
}

// B fragment 32x16 bf16 (KxN) from row-major f32 W[Hout][ldk]:
// B[k][n] = W[n][kbase+k]; lane = (kh = lane>>4, n = lane&15),
// lane supplies k = kh*8 .. kh*8+7 and 16 + kh*8 .. +7.
static __device__ __forceinline__ v16bf load_b_w(const float* __restrict__ W, int ldk,
                                                 int nGlobal, int kbase, int kh) {
  const float* p = W + (size_t)nGlobal * ldk + kbase + kh * 8;
  v4f b0 = ld4(p), b1 = ld4(p + 4), b2 = ld4(p + 16), b3 = ld4(p + 20);
  v16bf b;
  pack8(b, 0, b0, b1);
  pack8(b, 8, b2, b3);
  return b;
}

static __device__ __forceinline__ v8f wmma_bf16(v16bf a, v16bf b, v8f c) {
  // 8-arg form: (neg_a, A, neg_b, B, c_mod, C, reuse_a, reuse_b)
  return __builtin_amdgcn_wmma_f32_16x16x32_bf16(false, a, false, b, (short)0, c,
                                                 false, false);
}

static __device__ __forceinline__ float lrelu(float x) { return x > 0.f ? x : SLOPE * x; }
static __device__ __forceinline__ float sigmoidf_(float x) { return 1.f / (1.f + __expf(-x)); }
static __device__ __forceinline__ float softplusf_(float x) {
  return x > 20.f ? x : __logf(1.f + __expf(x));
}

// ---------------------------------------------------------------------------
// Fused Linear([*,K] -> [*,32]) + BN(eval) + LeakyReLU.
// Streams X with non-temporal loads. Writes f32 (if outF) and/or packed bf16
// (if outB). One wave per 16-row tile, both 16-col N-tiles.
// ---------------------------------------------------------------------------
__global__ __launch_bounds__(256) void embed_bn_lrelu_kernel(
    const float* __restrict__ X, long long M, int K,
    const float* __restrict__ W, const float* __restrict__ bias,
    const float* __restrict__ g, const float* __restrict__ be,
    const float* __restrict__ mean, const float* __restrict__ var,
    float* __restrict__ outF, __bf16* __restrict__ outB) {
  const int lane = threadIdx.x & 31;
  const int hf = lane >> 4;
  const int nl = lane & 15;
  const long long wave = (long long)((blockIdx.x * blockDim.x + threadIdx.x) >> 5);
  const long long nWaves = ((long long)gridDim.x * blockDim.x) >> 5;
  const long long tiles = (M + 15) >> 4;

  // fold bias+BN: y = acc*sc + sh, per output column (this lane: nl, nl+16)
  const int c0i = nl, c1i = nl + 16;
  const float sc0 = g[c0i] * rsqrtf(var[c0i] + EPSV);
  const float sh0 = be[c0i] + (bias[c0i] - mean[c0i]) * sc0;
  const float sc1 = g[c1i] * rsqrtf(var[c1i] + EPSV);
  const float sh1 = be[c1i] + (bias[c1i] - mean[c1i]) * sc1;

  for (long long tile = wave; tile < tiles; tile += nWaves) {
    const long long rowBase = tile << 4;
    long long m = rowBase + nl;
    if (m >= M) m = M - 1;  // clamp reads; writes guarded below
    const float* row = X + m * (long long)K;

    v8f acc0 = {}, acc1 = {};
    for (int kb = 0; kb < K; kb += 32) {
      v16bf a  = load_a_row<true>(row, kb, hf);
      v16bf b0 = load_b_w(W, K, nl,      kb, hf);
      v16bf b1 = load_b_w(W, K, nl + 16, kb, hf);
      acc0 = wmma_bf16(a, b0, acc0);
      acc1 = wmma_bf16(a, b1, acc1);
    }
#pragma unroll
    for (int r = 0; r < 8; ++r) {  // C layout: row = r + 8*hf, col = nl (+16)
      const long long orow = rowBase + r + hf * 8;
      if (orow < M) {
        const float y0 = lrelu(acc0[r] * sc0 + sh0);
        const float y1 = lrelu(acc1[r] * sc1 + sh1);
        if (outF) {
          outF[orow * HID + nl]      = y0;
          outF[orow * HID + nl + 16] = y1;
        }
        if (outB) {
          outB[orow * HID + nl]      = (__bf16)y0;
          outB[orow * HID + nl + 16] = (__bf16)y1;
        }
      }
    }
  }
}

// ---------------------------------------------------------------------------
// f32 -> packed bf16 mirror (8 elements per thread).
// ---------------------------------------------------------------------------
__global__ __launch_bounds__(256) void f32_to_bf16_kernel(
    const float* __restrict__ in, __bf16* __restrict__ out, long long n) {
  const long long stride = (long long)gridDim.x * blockDim.x * 8;
  for (long long i = ((long long)blockIdx.x * blockDim.x + threadIdx.x) * 8;
       i + 7 < n; i += stride) {
    v4f a = ld4(in + i), b = ld4(in + i + 4);
    v8bf o;
#pragma unroll
    for (int j = 0; j < 4; ++j) { o[j] = (__bf16)a[j]; o[4 + j] = (__bf16)b[j]; }
    *(v8bf*)(out + i) = o;
  }
}

// ---------------------------------------------------------------------------
// One CGConv layer. z = [h[dst] | h[src] | ea], K = 96 = 3x32, all bf16-packed.
// 12 weight B-fragments persist in VGPRs; scatter via global_atomic_add_f32.
// ---------------------------------------------------------------------------
__global__ __launch_bounds__(256) void cgconv_kernel(
    const __bf16* __restrict__ hb, const __bf16* __restrict__ ea,
    const int* __restrict__ srcIdx, const int* __restrict__ dstIdx,
    const float* __restrict__ Wf, const float* __restrict__ bFv,
    const float* __restrict__ Ws, const float* __restrict__ bSv,
    float* __restrict__ hnext, long long E) {
  const int lane = threadIdx.x & 31;
  const int hf = lane >> 4;
  const int nl = lane & 15;
  const long long wave = (long long)((blockIdx.x * blockDim.x + threadIdx.x) >> 5);
  const long long nWaves = ((long long)gridDim.x * blockDim.x) >> 5;
  const long long tiles = (E + 15) >> 4;

  v16bf wf_[2][3], ws_[2][3];
#pragma unroll
  for (int t = 0; t < 2; ++t)
#pragma unroll
    for (int c = 0; c < 3; ++c) {
      wf_[t][c] = load_b_w(Wf, 96, nl + 16 * t, 32 * c, hf);
      ws_[t][c] = load_b_w(Ws, 96, nl + 16 * t, 32 * c, hf);
    }
  const float bf0 = bFv[nl], bf1 = bFv[nl + 16];
  const float bs0 = bSv[nl], bs1 = bSv[nl + 16];

  for (long long tile = wave; tile < tiles; tile += nWaves) {
    const long long eBase = tile << 4;
    long long e = eBase + nl;
    if (e >= E) e = E - 1;
    const int di = dstIdx[e];
    const int si = srcIdx[e];

    // gathered bf16 A fragments (h mirror + ea are L2-resident)
    v16bf aD = load_a_row_bf16(hb + (size_t)di * HID, hf);  // z chunk 0: x_i
    v16bf aS = load_a_row_bf16(hb + (size_t)si * HID, hf);  // z chunk 1: x_j
    v16bf aE = load_a_row_bf16(ea + (size_t)e  * HID, hf);  // z chunk 2: e_ij

    v8f cf0 = {}, cf1 = {}, cs0 = {}, cs1 = {};
    cf0 = wmma_bf16(aD, wf_[0][0], cf0);  cf1 = wmma_bf16(aD, wf_[1][0], cf1);
    cs0 = wmma_bf16(aD, ws_[0][0], cs0);  cs1 = wmma_bf16(aD, ws_[1][0], cs1);
    cf0 = wmma_bf16(aS, wf_[0][1], cf0);  cf1 = wmma_bf16(aS, wf_[1][1], cf1);
    cs0 = wmma_bf16(aS, ws_[0][1], cs0);  cs1 = wmma_bf16(aS, ws_[1][1], cs1);
    cf0 = wmma_bf16(aE, wf_[0][2], cf0);  cf1 = wmma_bf16(aE, wf_[1][2], cf1);
    cs0 = wmma_bf16(aE, ws_[0][2], cs0);  cs1 = wmma_bf16(aE, ws_[1][2], cs1);

#pragma unroll
    for (int r = 0; r < 8; ++r) {  // C row = r + 8*hf -> edge eBase + that
      const long long ee = eBase + r + hf * 8;
      // dst[eBase + j] is held as `di` by lane j (and j+16): share via shuffle
      const int d = __shfl(di, r + 8 * hf, 32);  // ds_bpermute_b32
      if (ee < E) {
        const float m0 = sigmoidf_(cf0[r] + bf0) * softplusf_(cs0[r] + bs0);
        const float m1 = sigmoidf_(cf1[r] + bf1) * softplusf_(cs1[r] + bs1);
        atomicAdd(&hnext[(size_t)d * HID + nl],      m0);
        atomicAdd(&hnext[(size_t)d * HID + nl + 16], m1);
      }
    }
  }
}

// ---------------------------------------------------------------------------
// Output head: Linear([N,32] -> [N,128]) + BN + LeakyReLU from the bf16 mirror.
// K = 32 (one chunk); blockIdx.y picks the 32-wide column group.
// ---------------------------------------------------------------------------
__global__ __launch_bounds__(256) void out_bn_lrelu_kernel(
    const __bf16* __restrict__ Hb, long long M,
    const float* __restrict__ W,  // [128, 32] row-major
    const float* __restrict__ bias,
    const float* __restrict__ g, const float* __restrict__ be,
    const float* __restrict__ mean, const float* __restrict__ var,
    float* __restrict__ out) {
  const int lane = threadIdx.x & 31;
  const int hf = lane >> 4;
  const int nl = lane & 15;
  const int colBase = blockIdx.y * 32;
  const long long wave = (long long)((blockIdx.x * blockDim.x + threadIdx.x) >> 5);
  const long long nWaves = ((long long)gridDim.x * blockDim.x) >> 5;
  const long long tiles = (M + 15) >> 4;

  const int c0i = colBase + nl, c1i = colBase + nl + 16;
  const float sc0 = g[c0i] * rsqrtf(var[c0i] + EPSV);
  const float sh0 = be[c0i] + (bias[c0i] - mean[c0i]) * sc0;
  const float sc1 = g[c1i] * rsqrtf(var[c1i] + EPSV);
  const float sh1 = be[c1i] + (bias[c1i] - mean[c1i]) * sc1;

  const v16bf b0 = load_b_w(W, HID, c0i, 0, hf);
  const v16bf b1 = load_b_w(W, HID, c1i, 0, hf);

  for (long long tile = wave; tile < tiles; tile += nWaves) {
    const long long rowBase = tile << 4;
    long long m = rowBase + nl;
    if (m >= M) m = M - 1;
    v16bf a = load_a_row_bf16(Hb + m * HID, hf);
    v8f acc0 = {}, acc1 = {};
    acc0 = wmma_bf16(a, b0, acc0);
    acc1 = wmma_bf16(a, b1, acc1);
#pragma unroll
    for (int r = 0; r < 8; ++r) {
      const long long orow = rowBase + r + hf * 8;
      if (orow < M) {
        out[orow * 128 + c0i] = lrelu(acc0[r] * sc0 + sh0);
        out[orow * 128 + c1i] = lrelu(acc1[r] * sc1 + sh1);
      }
    }
  }
}

// ---------------------------------------------------------------------------
extern "C" void kernel_launch(void* const* d_in, const int* in_sizes, int n_in,
                              void* d_out, int out_size, void* d_ws, size_t ws_size,
                              hipStream_t stream) {
  const float* x          = (const float*)d_in[0];
  const int*   edge_index = (const int*)  d_in[1];
  const float* edge_attr  = (const float*)d_in[2];
  const float* w_in  = (const float*)d_in[3];
  const float* b_in  = (const float*)d_in[4];
  const float* g_in  = (const float*)d_in[5];
  const float* be_in = (const float*)d_in[6];
  const float* m_in  = (const float*)d_in[7];
  const float* v_in  = (const float*)d_in[8];
  const float* w_e   = (const float*)d_in[9];
  const float* b_e   = (const float*)d_in[10];
  const float* g_e   = (const float*)d_in[11];
  const float* be_e  = (const float*)d_in[12];
  const float* m_e   = (const float*)d_in[13];
  const float* v_e   = (const float*)d_in[14];
  const float* wf    = (const float*)d_in[15];  // [L,32,96]
  const float* bfv   = (const float*)d_in[16];  // [L,32]
  const float* wsm   = (const float*)d_in[17];  // [L,32,96]
  const float* bsv   = (const float*)d_in[18];  // [L,32]
  const float* w_out = (const float*)d_in[19];  // [128,32]
  const float* b_out = (const float*)d_in[20];
  const float* g_out = (const float*)d_in[21];
  const float* be_out= (const float*)d_in[22];
  const float* m_out = (const float*)d_in[23];
  const float* v_out = (const float*)d_in[24];

  const long long N = in_sizes[0] / 128;
  const long long E = in_sizes[2] / 128;
  const int* srcIdx = edge_index;            // edge_index[0] = src (j)
  const int* dstIdx = edge_index + E;        // edge_index[1] = dst (i)

  // workspace: ea_bf16 [E,32] | hA f32 [N,32] | hB f32 [N,32] | hb16 [N,32]
  char* wsb = (char*)d_ws;
  const size_t eaBytes = (((size_t)E * HID * sizeof(__bf16)) + 255) & ~(size_t)255;
  const size_t hBytes  = (((size_t)N * HID * sizeof(float))  + 255) & ~(size_t)255;
  __bf16* ea   = (__bf16*)wsb;
  float*  hA   = (float*)(wsb + eaBytes);
  float*  hB   = (float*)(wsb + eaBytes + hBytes);
  __bf16* hb16 = (__bf16*)(wsb + eaBytes + 2 * hBytes);
  (void)n_in; (void)out_size; (void)ws_size;

  const dim3 blk(256);  // 8 waves/block
  const long long nTiles = (N + 15) / 16;
  const long long eTiles = (E + 15) / 16;
  int nBlocks = (int)((nTiles + 7) / 8); if (nBlocks > 4096) nBlocks = 4096;
  int eBlocks = (int)((eTiles + 7) / 8); if (eBlocks > 4096) eBlocks = 4096;
  const long long hElems = N * HID;
  int cBlocks = (int)((hElems / 8 + 255) / 256); if (cBlocks > 2048) cBlocks = 2048;

  // 1) node embed -> hA (f32) + hb16 mirror; edge embed -> ea (bf16 only)
  embed_bn_lrelu_kernel<<<nBlocks, blk, 0, stream>>>(x, N, 128, w_in, b_in,
                                                     g_in, be_in, m_in, v_in,
                                                     hA, hb16);
  embed_bn_lrelu_kernel<<<eBlocks, blk, 0, stream>>>(edge_attr, E, 128, w_e, b_e,
                                                     g_e, be_e, m_e, v_e,
                                                     (float*)nullptr, ea);

  // 2) layer 0: hB = hA + scatter(msg(hb16, ea))
  hipMemcpyAsync(hB, hA, (size_t)N * HID * sizeof(float),
                 hipMemcpyDeviceToDevice, stream);
  cgconv_kernel<<<eBlocks, blk, 0, stream>>>(hb16, ea, srcIdx, dstIdx,
                                             wf, bfv, wsm, bsv, hB, E);
  f32_to_bf16_kernel<<<cBlocks, blk, 0, stream>>>(hB, hb16, hElems);

  // 3) layer 1: hA = hB + scatter(msg(hb16, ea))
  hipMemcpyAsync(hA, hB, (size_t)N * HID * sizeof(float),
                 hipMemcpyDeviceToDevice, stream);
  cgconv_kernel<<<eBlocks, blk, 0, stream>>>(hb16, ea, srcIdx, dstIdx,
                                             wf + 32 * 96, bfv + 32,
                                             wsm + 32 * 96, bsv + 32, hA, E);
  f32_to_bf16_kernel<<<cBlocks, blk, 0, stream>>>(hA, hb16, hElems);

  // 4) output head from the bf16 mirror
  dim3 og((unsigned)((nTiles + 7) / 8), 4);
  out_bn_lrelu_kernel<<<og, blk, 0, stream>>>(hb16, N, w_out, b_out,
                                              g_out, be_out, m_out, v_out,
                                              (float*)d_out);
}
